// Encoder_18949395710485
// MI455X (gfx1250) — matrix operand order repaired and checked by reading, not compiled, not run on hardware
//
#include <hip/hip_runtime.h>

#define BB   4
#define SS   2048
#define DD   512
#define HH   8
#define DKK  64
#define NLL  6
#define DFFF 2048

typedef __attribute__((ext_vector_type(16))) __bf16 v16bf;
typedef __attribute__((ext_vector_type(8)))  __bf16 v8bf;
typedef __attribute__((ext_vector_type(8)))  float  v8f;
typedef __attribute__((ext_vector_type(4)))  unsigned v4u;
typedef __attribute__((ext_vector_type(4)))  int    v4i;
typedef __attribute__((ext_vector_type(8)))  int    v8i;

static __device__ __forceinline__ __bf16 f2bf(float f) {
  unsigned int x = __builtin_bit_cast(unsigned int, f);
  x += 0x7FFFu + ((x >> 16) & 1u);           // round-to-nearest-even
  unsigned short h = (unsigned short)(x >> 16);
  return __builtin_bit_cast(__bf16, h);
}

static __device__ __forceinline__ v8f wmma_bf16(v16bf a, v16bf b, v8f c) {
  return __builtin_amdgcn_wmma_f32_16x16x32_bf16(false, a, false, b, (short)0, c,
                                                 false, false);
}

// ---------------------------------------------------------------------------
// TDM: async 2-D tile load (bf16 elements) global -> LDS.
// tile0 = contiguous elements per row, tile1 = rows, stride = row stride (elems)
// D# layout per CDNA5 ISA ch.8 (group0 128b, group1 256b; groups 2/3 zero = 2D)
// ---------------------------------------------------------------------------
static __device__ __forceinline__ void tdm_load_2d(unsigned lds_addr,
                                                   const void* gptr,
                                                   unsigned tile0, unsigned tile1,
                                                   unsigned stride) {
  unsigned long long ga = (unsigned long long)gptr;
  v4u g0;
  g0[0] = 1u;                                          // count=1 (user D#)
  g0[1] = lds_addr;                                    // lds_addr [63:32]
  g0[2] = (unsigned)(ga & 0xFFFFFFFFu);                // global_addr [95:64]
  g0[3] = (unsigned)((ga >> 32) & 0x01FFFFFFu) | 0x80000000u;  // addr hi | type=2
  v8i g1;
  g1[0] = (int)(1u << 16);                             // data_size=1 -> 2 bytes
  g1[1] = (int)(tile0 << 16);                          // tensor_dim0 lo16
  g1[2] = (int)((tile0 >> 16) | ((tile1 & 0xFFFFu) << 16)); // dim0 hi | dim1 lo
  g1[3] = (int)((tile1 >> 16) | (tile0 << 16));        // dim1 hi | tile_dim0
  g1[4] = (int)(tile1 & 0xFFFFu);                      // tile_dim1 (tile_dim2=0)
  g1[5] = (int)stride;                                 // tensor_dim0_stride lo
  g1[6] = 0;
  g1[7] = 0;
  v4i z4 = {0, 0, 0, 0};
#if __clang_major__ >= 23
  v8i z8 = {0, 0, 0, 0, 0, 0, 0, 0};
  __builtin_amdgcn_tensor_load_to_lds(g0, g1, z4, z4, z8, 0);
#else
  __builtin_amdgcn_tensor_load_to_lds(g0, g1, z4, z4, 0);
#endif
}

// ---------------------------------------------------------------------------
// x[tok,d] = embed[src[tok],d]*sqrt(D) + PE(s,d)
// ---------------------------------------------------------------------------
__global__ __launch_bounds__(256)
void embed_kernel(const int* __restrict__ src, const float* __restrict__ embed,
                  float* __restrict__ x) {
  const int tok = blockIdx.x;            // b*SS + s
  const int s = tok & (SS - 1);
  const int v = src[tok];
  const float scale = 22.62741699796952f;   // sqrt(512)
  for (int d = threadIdx.x; d < DD; d += 256) {
    float freq = powf(10000.0f, -(float)d * (1.0f / 256.0f)); // 10000^(-2d/512)
    float ang  = (float)s * freq;
    float pe   = (d & 1) ? cosf(ang) : sinf(ang);
    x[(size_t)tok * DD + d] = embed[(size_t)v * DD + d] * scale + pe;
  }
}

// ---------------------------------------------------------------------------
// row LayerNorm over 512; optional f32 and/or bf16 outputs
// ---------------------------------------------------------------------------
__global__ __launch_bounds__(256)
void ln_kernel(const float* __restrict__ x, const float* __restrict__ g,
               const float* __restrict__ bt, float* __restrict__ outF,
               __bf16* __restrict__ outB) {
  const int row = blockIdx.x;
  const int tid = threadIdx.x;
  const float* xr = x + (size_t)row * DD;
  float v0 = xr[tid], v1 = xr[tid + 256];
  float s = v0 + v1, q = v0 * v0 + v1 * v1;
#pragma unroll
  for (int m = 1; m < 32; m <<= 1) {
    s += __shfl_xor(s, m, 32);
    q += __shfl_xor(q, m, 32);
  }
  __shared__ float ws[8], wq[8];
  __shared__ float mean_s, rstd_s;
  if ((tid & 31) == 0) { ws[tid >> 5] = s; wq[tid >> 5] = q; }
  __syncthreads();
  if (tid == 0) {
    float ts = 0.f, tq = 0.f;
#pragma unroll
    for (int i = 0; i < 8; ++i) { ts += ws[i]; tq += wq[i]; }
    float mean = ts * (1.0f / 512.0f);
    float var  = tq * (1.0f / 512.0f) - mean * mean;
    mean_s = mean;
    rstd_s = rsqrtf(var + 1e-5f);
  }
  __syncthreads();
  const float mean = mean_s, rstd = rstd_s;
  float y0 = (v0 - mean) * rstd * g[tid] + bt[tid];
  float y1 = (v1 - mean) * rstd * g[tid + 256] + bt[tid + 256];
  if (outF) {
    outF[(size_t)row * DD + tid]       = y0;
    outF[(size_t)row * DD + tid + 256] = y1;
  }
  if (outB) {
    outB[(size_t)row * DD + tid]       = f2bf(y0);
    outB[(size_t)row * DD + tid + 256] = f2bf(y1);
  }
}

// ---------------------------------------------------------------------------
// W f32 [K][N] -> Wt bf16 [N][K]   (32x32 LDS tile transpose, one-time)
// ---------------------------------------------------------------------------
__global__ __launch_bounds__(256)
void transpose_w_kernel(const float* __restrict__ in, __bf16* __restrict__ out,
                        int K, int N) {
  __shared__ __bf16 tile[32][33];
  const int kx = blockIdx.y * 32, nx = blockIdx.x * 32;
  const int tx = threadIdx.x & 31, ty = threadIdx.x >> 5;   // 32 x 8
#pragma unroll
  for (int i = 0; i < 32; i += 8)
    tile[ty + i][tx] = f2bf(in[(size_t)(kx + ty + i) * N + nx + tx]);
  __syncthreads();
#pragma unroll
  for (int i = 0; i < 32; i += 8)
    out[(size_t)(nx + ty + i) * K + kx + tx] = tile[tx][ty + i];
}

// ---------------------------------------------------------------------------
// [B,S,D] bf16 -> [B,H,S,DK] bf16  (8 elements per thread, both sides dense)
// ---------------------------------------------------------------------------
__global__ __launch_bounds__(256)
void pack_qk_kernel(const __bf16* __restrict__ in, __bf16* __restrict__ out) {
  const int idx = blockIdx.x * 256 + threadIdx.x;   // TOK*D/8 = 524288
  const int base = idx * 8;
  const int d = base & (DD - 1);
  const int t = base >> 9;
  const int b = t >> 11;
  const int s = t & (SS - 1);
  const int h = d >> 6;
  const int dk = d & 63;
  *(v8bf*)&out[((((size_t)(b * HH + h)) * SS + s) << 6) + dk] =
      *(const v8bf*)&in[base];
}

// [B,S,D] bf16 -> [B,H,DK,S] bf16 (transposed V; 8-wide stores along S)
__global__ __launch_bounds__(256)
void pack_vt_kernel(const __bf16* __restrict__ in, __bf16* __restrict__ out) {
  const int idx = blockIdx.x * 256 + threadIdx.x;   // 2^19
  const int s0 = (idx & 255) * 8;
  const int dk = (idx >> 8) & 63;
  const int h  = (idx >> 14) & 7;
  const int b  = idx >> 17;
  v8bf v;
#pragma unroll
  for (int i = 0; i < 8; ++i)
    v[i] = in[((size_t)(b * SS + s0 + i)) * DD + h * DKK + dk];
  *(v8bf*)&out[(((size_t)(b * HH + h)) * DKK + dk) * SS + s0] = v;
}

// ---------------------------------------------------------------------------
// C[M,N] = act(A_bf16[M,K] @ Wt_bf16[N][K]^T + bias (+ resid))
// 64x64 tile / 128 threads / 4 waves; double-buffered TDM feeds the LDS tiles.
// ---------------------------------------------------------------------------
__global__ __launch_bounds__(128)
void gemm_bf16_kernel(const __bf16* __restrict__ A, const __bf16* __restrict__ Wt,
                      const float* __restrict__ bias,
                      const float* __restrict__ resid, float* __restrict__ outF,
                      __bf16* __restrict__ outB, int M, int N, int K, int relu) {
  __shared__ __bf16 sA[2][64][32];     // [buf][m][k]
  __shared__ __bf16 sBT[2][64][32];    // [buf][n][k]
  // Escape the LDS arrays (TDM writes them behind the compiler's back), but
  // keep all reads on the direct array path so they stay DS (addrspace 3) ops.
  {
    void* eA = (void*)&sA[0][0][0];
    void* eB = (void*)&sBT[0][0][0];
    asm volatile("" ::"v"(eA), "v"(eB) : "memory");
  }
  const unsigned ldsA = (unsigned)(unsigned long long)(void*)&sA[0][0][0];
  const unsigned ldsB = (unsigned)(unsigned long long)(void*)&sBT[0][0][0];

  const int tid = threadIdx.x;
  const int lane = tid & 31;
  const int lh = lane >> 4, ll = lane & 15;
  const int wid = tid >> 5;
  const int m0 = blockIdx.y * 64;
  const int n0 = blockIdx.x * 64;
  const int wm = (wid >> 1) * 32;
  const int wn = (wid & 1) * 32;
  v8f c00 = {}, c01 = {}, c10 = {}, c11 = {};

  if (tid < 32) {                        // prologue: fetch tile pair 0
    tdm_load_2d(ldsA, &A[(size_t)m0 * K], 32u, 64u, (unsigned)K);
    tdm_load_2d(ldsB, &Wt[(size_t)n0 * K], 32u, 64u, (unsigned)K);
  }

  for (int k0 = 0; k0 < K; k0 += 32) {
    const int cur = (k0 >> 5) & 1;
    if (tid < 32) {                      // wave 0 drives the tensor DMA
      if (k0 + 32 < K) {
        const unsigned boff = (unsigned)((cur ^ 1) * 64 * 32 * 2);
        tdm_load_2d(ldsA + boff, &A[(size_t)m0 * K + k0 + 32], 32u, 64u,
                    (unsigned)K);
        tdm_load_2d(ldsB + boff, &Wt[(size_t)n0 * K + k0 + 32], 32u, 64u,
                    (unsigned)K);
        __builtin_amdgcn_s_wait_tensorcnt(2);  // current pair resident
      } else {
        __builtin_amdgcn_s_wait_tensorcnt(0);
      }
    }
    __syncthreads();                     // buf[cur] ready for everyone
    asm volatile("" ::: "memory");

    v16bf a0, a1, b0, b1;
    {
      const int ka = lh * 16;            // A frag: K = lh*16 + e
      v8bf a0l = *(const v8bf*)&sA[cur][wm + ll][ka];
      v8bf a0h = *(const v8bf*)&sA[cur][wm + ll][ka + 8];
      v8bf a1l = *(const v8bf*)&sA[cur][wm + 16 + ll][ka];
      v8bf a1h = *(const v8bf*)&sA[cur][wm + 16 + ll][ka + 8];
      a0 = __builtin_shufflevector(a0l, a0h, 0, 1, 2, 3, 4, 5, 6, 7, 8, 9, 10,
                                   11, 12, 13, 14, 15);
      a1 = __builtin_shufflevector(a1l, a1h, 0, 1, 2, 3, 4, 5, 6, 7, 8, 9, 10,
                                   11, 12, 13, 14, 15);
      const int kb = lh * 16;            // B frag: K = lh*16 + e
      b0 = *(const v16bf*)&sBT[cur][wn + ll][kb];
      b1 = *(const v16bf*)&sBT[cur][wn + 16 + ll][kb];
    }
    c00 = wmma_bf16(a0, b0, c00);
    c01 = wmma_bf16(a0, b1, c01);
    c10 = wmma_bf16(a1, b0, c10);
    c11 = wmma_bf16(a1, b1, c11);
    __syncthreads();                     // done reading buf[cur] before reuse
  }

  const int gn0 = n0 + wn + ll;
  const int gn1 = gn0 + 16;
  const float bia0 = bias[gn0];
  const float bia1 = bias[gn1];
#pragma unroll
  for (int r = 0; r < 8; ++r) {          // C row = r + 8*lh
    int gm0 = m0 + wm + lh * 8 + r;
    int gm1 = gm0 + 16;
    float v00 = c00[r] + bia0, v01 = c01[r] + bia1;
    float v10 = c10[r] + bia0, v11 = c11[r] + bia1;
    if (resid) {
      v00 += resid[(size_t)gm0 * N + gn0];
      v01 += resid[(size_t)gm0 * N + gn1];
      v10 += resid[(size_t)gm1 * N + gn0];
      v11 += resid[(size_t)gm1 * N + gn1];
    }
    if (relu) {
      v00 = fmaxf(v00, 0.f); v01 = fmaxf(v01, 0.f);
      v10 = fmaxf(v10, 0.f); v11 = fmaxf(v11, 0.f);
    }
    if (outF) {
      outF[(size_t)gm0 * N + gn0] = v00;
      outF[(size_t)gm0 * N + gn1] = v01;
      outF[(size_t)gm1 * N + gn0] = v10;
      outF[(size_t)gm1 * N + gn1] = v11;
    }
    if (outB) {
      outB[(size_t)gm0 * N + gn0] = f2bf(v00);
      outB[(size_t)gm0 * N + gn1] = f2bf(v01);
      outB[(size_t)gm1 * N + gn0] = f2bf(v10);
      outB[(size_t)gm1 * N + gn1] = f2bf(v11);
    }
  }
}

// ---------------------------------------------------------------------------
// Flash attention: one wave per (b, h, 16-query tile); online softmax.
// Qh,Kh: [B,H,S,DK] bf16 ; Vt: [B,H,DK,S] bf16 ; out: [B,S,D] bf16
// ---------------------------------------------------------------------------
__global__ __launch_bounds__(32)
void attn_kernel(const __bf16* __restrict__ Qh, const __bf16* __restrict__ Kh,
                 const __bf16* __restrict__ Vt, const int* __restrict__ mask,
                 __bf16* __restrict__ outB) {
  const int lane = threadIdx.x;
  const int q0 = blockIdx.x * 16;
  const int h = blockIdx.y;
  const int b = blockIdx.z;
  const size_t head = ((size_t)(b * HH + h)) * SS * DKK;
  const __bf16* Q = Qh + head;
  const __bf16* K = Kh + head;
  const __bf16* V = Vt + head;                 // [DKK][SS]
  const int* mrow = mask + b * SS;

  __shared__ __bf16 sP[16][32];

  const int lh = lane >> 4;
  const int ll = lane & 15;

  v16bf qa0, qa1;                              // Q 16x64 as two 16x32 A frags
  {
    const __bf16* qr = Q + (size_t)(q0 + ll) * DKK;
    const int ka = lh * 16;
    v8bf l0 = *(const v8bf*)(qr + ka);
    v8bf h0 = *(const v8bf*)(qr + 8 + ka);
    v8bf l1 = *(const v8bf*)(qr + 32 + ka);
    v8bf h1 = *(const v8bf*)(qr + 40 + ka);
    qa0 = __builtin_shufflevector(l0, h0, 0, 1, 2, 3, 4, 5, 6, 7, 8, 9, 10, 11,
                                  12, 13, 14, 15);
    qa1 = __builtin_shufflevector(l1, h1, 0, 1, 2, 3, 4, 5, 6, 7, 8, 9, 10, 11,
                                  12, 13, 14, 15);
  }

  float rm[8], rl[8];
  v8f o0 = {}, o1 = {}, o2 = {}, o3 = {};
#pragma unroll
  for (int r = 0; r < 8; ++r) { rm[r] = -__builtin_inff(); rl[r] = 0.0f; }

  const float sc = 0.125f;                     // 1/sqrt(64)

  for (int c0 = 0; c0 < SS; c0 += 32) {
    v8f s0 = {}, s1 = {};
    {
      const __bf16* kr = K + (size_t)(c0 + ll) * DKK;
      const int db = lh * 16;
      v16bf kb0 = *(const v16bf*)(kr + db);
      v16bf kb1 = *(const v16bf*)(kr + 32 + db);
      s0 = wmma_bf16(qa0, kb0, s0);
      s0 = wmma_bf16(qa1, kb1, s0);
    }
    {
      const __bf16* kr = K + (size_t)(c0 + 16 + ll) * DKK;
      const int db = lh * 16;
      v16bf kb0 = *(const v16bf*)(kr + db);
      v16bf kb1 = *(const v16bf*)(kr + 32 + db);
      s1 = wmma_bf16(qa0, kb0, s1);
      s1 = wmma_bf16(qa1, kb1, s1);
    }
    const bool ok0 = mrow[c0 + ll] != 0;
    const bool ok1 = mrow[c0 + 16 + ll] != 0;
    float tmax[8];
#pragma unroll
    for (int r = 0; r < 8; ++r) {
      float x0 = ok0 ? s0[r] * sc : -1.0e9f;
      float x1 = ok1 ? s1[r] * sc : -1.0e9f;
      s0[r] = x0; s1[r] = x1;
      tmax[r] = fmaxf(x0, x1);
    }
#pragma unroll
    for (int m = 1; m < 16; m <<= 1)           // row max within 16-lane group
#pragma unroll
      for (int r = 0; r < 8; ++r) tmax[r] = fmaxf(tmax[r], __shfl_xor(tmax[r], m, 32));

    float fr[8], ps[8];
#pragma unroll
    for (int r = 0; r < 8; ++r) {
      float nm = fmaxf(rm[r], tmax[r]);
      fr[r] = expf(rm[r] - nm);
      rm[r] = nm;
      float p0 = expf(s0[r] - nm);
      float p1 = expf(s1[r] - nm);
      s0[r] = p0; s1[r] = p1;
      ps[r] = p0 + p1;
    }
#pragma unroll
    for (int m = 1; m < 16; m <<= 1)           // row sum within 16-lane group
#pragma unroll
      for (int r = 0; r < 8; ++r) ps[r] += __shfl_xor(ps[r], m, 32);
#pragma unroll
    for (int r = 0; r < 8; ++r) {
      rl[r] = rl[r] * fr[r] + ps[r];
      o0[r] *= fr[r]; o1[r] *= fr[r]; o2[r] *= fr[r]; o3[r] *= fr[r];
    }
    // P: C-layout -> LDS -> A-layout
#pragma unroll
    for (int r = 0; r < 8; ++r) {
      sP[lh * 8 + r][ll]      = f2bf(s0[r]);
      sP[lh * 8 + r][16 + ll] = f2bf(s1[r]);
    }
    __syncthreads();
    v16bf pa;
    {
      const int ka = lh * 16;
      v8bf plo = *(const v8bf*)&sP[ll][ka];
      v8bf phi = *(const v8bf*)&sP[ll][ka + 8];
      pa = __builtin_shufflevector(plo, phi, 0, 1, 2, 3, 4, 5, 6, 7, 8, 9, 10,
                                   11, 12, 13, 14, 15);
    }
    {
      const int kv = c0 + lh * 16;
      v16bf vb0 = *(const v16bf*)(V + (size_t)(ll) * SS + kv);
      v16bf vb1 = *(const v16bf*)(V + (size_t)(16 + ll) * SS + kv);
      v16bf vb2 = *(const v16bf*)(V + (size_t)(32 + ll) * SS + kv);
      v16bf vb3 = *(const v16bf*)(V + (size_t)(48 + ll) * SS + kv);
      o0 = wmma_bf16(pa, vb0, o0);
      o1 = wmma_bf16(pa, vb1, o1);
      o2 = wmma_bf16(pa, vb2, o2);
      o3 = wmma_bf16(pa, vb3, o3);
    }
    __syncthreads();
  }
#pragma unroll
  for (int r = 0; r < 8; ++r) {
    const int qrow = q0 + lh * 8 + r;
    const float inv = 1.0f / rl[r];
    const size_t base = ((size_t)(b * SS + qrow)) * DD + h * DKK;
    outB[base + ll]      = f2bf(o0[r] * inv);
    outB[base + 16 + ll] = f2bf(o1[r] * inv);
    outB[base + 32 + ll] = f2bf(o2[r] * inv);
    outB[base + 48 + ll] = f2bf(o3[r] * inv);
  }
}

// ---------------------------------------------------------------------------
extern "C" void kernel_launch(void* const* d_in, const int* in_sizes, int n_in,
                              void* d_out, int out_size, void* d_ws,
                              size_t ws_size, hipStream_t stream) {
  (void)in_sizes; (void)n_in; (void)out_size; (void)ws_size;
  const int*   src   = (const int*)d_in[0];
  const int*   mask  = (const int*)d_in[1];
  const float* embed = (const float*)d_in[2];
  const float* ln1_g = (const float*)d_in[3];
  const float* ln1_b = (const float*)d_in[4];
  const float* Wq    = (const float*)d_in[5];
  const float* bq    = (const float*)d_in[6];
  const float* Wk    = (const float*)d_in[7];
  const float* bk    = (const float*)d_in[8];
  const float* Wv    = (const float*)d_in[9];
  const float* bv    = (const float*)d_in[10];
  const float* Wo    = (const float*)d_in[11];
  const float* bo    = (const float*)d_in[12];
  const float* ln2_g = (const float*)d_in[13];
  const float* ln2_b = (const float*)d_in[14];
  const float* W1    = (const float*)d_in[15];
  const float* b1    = (const float*)d_in[16];
  const float* W2    = (const float*)d_in[17];
  const float* b2    = (const float*)d_in[18];
  const float* lnf_g = (const float*)d_in[19];
  const float* lnf_b = (const float*)d_in[20];
  float* out = (float*)d_out;

  const size_t TOK = (size_t)BB * SS;                 // 8192
  const size_t ACT_F32 = TOK * DD * sizeof(float);    // 16 MB
  const size_t ACT_BF  = TOK * DD * 2;                // 8 MB
  const size_t WBLK    = 4 * DD * DD + 2 * (size_t)DD * DFFF;  // elems/layer

  char* ws = (char*)d_ws;
  size_t off = 0;
  auto carve = [&](size_t bytes) {
    void* r = ws + off;
    off += (bytes + 255) & ~(size_t)255;
    return r;
  };
  float*  x    = (float*)carve(ACT_F32);
  __bf16* xn   = (__bf16*)carve(ACT_BF);
  char*   regA = (char*)carve(4 * ACT_BF);     // qb|kb|vb|qh, aliased by hidden
  __bf16* qb   = (__bf16*)regA;
  __bf16* kb   = (__bf16*)(regA + ACT_BF);
  __bf16* vb   = (__bf16*)(regA + 2 * ACT_BF);
  __bf16* qh   = (__bf16*)(regA + 3 * ACT_BF);
  __bf16* hid  = (__bf16*)regA;                // alias (FFN phase only)
  __bf16* kh   = (__bf16*)carve(ACT_BF);
  __bf16* vt   = (__bf16*)carve(ACT_BF);
  __bf16* attn = (__bf16*)carve(ACT_BF);
  __bf16* wt   = (__bf16*)carve(WBLK * NLL * 2);

  // one-time weight convert+transpose to bf16 [N][K]
  const dim3 tDD(DD / 32, DD / 32);        // (16,16)
  const dim3 tW1(DFFF / 32, DD / 32);      // N=2048,K=512
  const dim3 tW2(DD / 32, DFFF / 32);      // N=512, K=2048
  for (int l = 0; l < NLL; ++l) {
    __bf16* wl = wt + (size_t)l * WBLK;
    transpose_w_kernel<<<tDD, 256, 0, stream>>>(Wq + (size_t)l * DD * DD,
                                                wl + 0 * DD * DD, DD, DD);
    transpose_w_kernel<<<tDD, 256, 0, stream>>>(Wk + (size_t)l * DD * DD,
                                                wl + 1 * DD * DD, DD, DD);
    transpose_w_kernel<<<tDD, 256, 0, stream>>>(Wv + (size_t)l * DD * DD,
                                                wl + 2 * DD * DD, DD, DD);
    transpose_w_kernel<<<tDD, 256, 0, stream>>>(Wo + (size_t)l * DD * DD,
                                                wl + 3 * DD * DD, DD, DD);
    transpose_w_kernel<<<tW1, 256, 0, stream>>>(W1 + (size_t)l * DD * DFFF,
                                                wl + 4 * DD * DD, DD, DFFF);
    transpose_w_kernel<<<tW2, 256, 0, stream>>>(W2 + (size_t)l * DFFF * DD,
                                                wl + 4 * DD * DD + (size_t)DD * DFFF,
                                                DFFF, DD);
  }

  embed_kernel<<<(int)TOK, 256, 0, stream>>>(src, embed, x);

  const dim3 gD(DD / 64, (int)TOK / 64);       // (8,128) for N=512
  const dim3 gF(DFFF / 64, (int)TOK / 64);     // (32,128) for N=2048
  const int packQK = (int)(TOK * DD / 8 / 256);    // 2048 blocks
  const int packVT = (int)(TOK * DD / 8 / 256);
  const dim3 gA(SS / 16, HH, BB);              // (128,8,4)

  for (int l = 0; l < NLL; ++l) {
    const __bf16* wl  = wt + (size_t)l * WBLK;
    const __bf16* wqT = wl + 0 * DD * DD;
    const __bf16* wkT = wl + 1 * DD * DD;
    const __bf16* wvT = wl + 2 * DD * DD;
    const __bf16* woT = wl + 3 * DD * DD;
    const __bf16* w1T = wl + 4 * DD * DD;                       // [DFF][D]
    const __bf16* w2T = wl + 4 * DD * DD + (size_t)DD * DFFF;   // [D][DFF]

    ln_kernel<<<(int)TOK, 256, 0, stream>>>(x, ln1_g + l * DD, ln1_b + l * DD,
                                            nullptr, xn);
    gemm_bf16_kernel<<<gD, 128, 0, stream>>>(xn, wqT, bq + l * DD, nullptr,
                                             nullptr, qb, (int)TOK, DD, DD, 0);
    gemm_bf16_kernel<<<gD, 128, 0, stream>>>(xn, wkT, bk + l * DD, nullptr,
                                             nullptr, kb, (int)TOK, DD, DD, 0);
    gemm_bf16_kernel<<<gD, 128, 0, stream>>>(xn, wvT, bv + l * DD, nullptr,
                                             nullptr, vb, (int)TOK, DD, DD, 0);
    pack_qk_kernel<<<packQK, 256, 0, stream>>>(qb, qh);
    pack_qk_kernel<<<packQK, 256, 0, stream>>>(kb, kh);
    pack_vt_kernel<<<packVT, 256, 0, stream>>>(vb, vt);
    attn_kernel<<<gA, 32, 0, stream>>>(qh, kh, vt, mask, attn);
    gemm_bf16_kernel<<<gD, 128, 0, stream>>>(attn, woT, bo + l * DD, x, x,
                                             nullptr, (int)TOK, DD, DD, 0);
    ln_kernel<<<(int)TOK, 256, 0, stream>>>(x, ln2_g + l * DD, ln2_b + l * DD,
                                            nullptr, xn);
    gemm_bf16_kernel<<<gF, 128, 0, stream>>>(xn, w1T, b1 + l * DFFF, nullptr,
                                             nullptr, hid, (int)TOK, DFFF, DD, 1);
    gemm_bf16_kernel<<<gD, 128, 0, stream>>>(hid, w2T, b2 + l * DD, x, x,
                                             nullptr, (int)TOK, DD, DFFF, 0);
  }

  ln_kernel<<<(int)TOK, 256, 0, stream>>>(x, lnf_g, lnf_b, out, nullptr);
}
